// BiMamba3Block_16406775071290
// MI455X (gfx1250) — compile-verified, matrix operand order checked by reading, not run on hardware
//
#include <hip/hip_runtime.h>

// ---------------- problem constants ----------------
#define DMODEL  256
#define DSTATE  32
#define HEADD   64
#define RANK    4
#define DINNER  512
#define NHEADS  8
#define CONVK   4
#define CONVCH  576            // DINNER + 2*DSTATE
#define NPROJ   1096           // 2*DINNER + 2*DSTATE + NHEADS
#define BNB     8
#define SEQ     2048
#define ROWS    (BNB*SEQ)      // 16384
#define NCHUNK  16
#define CLEN    (SEQ/NCHUNK)   // 128
#define MTILES  (ROWS/16)      // 1024

#define NTW     4              // n-tiles per wave in WMMA GEMMs
#define NTP_IN  72             // 1096 cols -> 69 tiles, padded to 72 (18 groups of 4)
#define KC_IN   (DMODEL/32)    // 8  k-chunks for in-proj
#define NT_OUT  16             // 256 cols -> 16 tiles (4 groups of 4)
#define KC_OUT  (DINNER/32)    // 16 k-chunks for out-proj

typedef __attribute__((ext_vector_type(16))) __bf16 v16bf;
typedef __attribute__((ext_vector_type(8)))  __bf16 v8bf;
typedef __attribute__((ext_vector_type(8)))  float  v8f;

struct Params {
  const float* Win;  const float* convw; const float* convb; const float* dtb;
  const float* Alog; const float* Aim;   const float* Dsk;
  const float* Wmri; const float* Wmro;  const float* normw; const float* Wout;
};

// ---------------- helpers ----------------
__device__ inline unsigned short f2bf(float f) {
  unsigned int u = __float_as_uint(f);
  unsigned int r = u + 0x7FFFu + ((u >> 16) & 1u);   // RNE
  return (unsigned short)(r >> 16);
}
__device__ inline float siluf(float x) { return x / (1.f + __expf(-x)); }

// trapezoidal (bilinear) discretization coefficients, complex A = Ar + i*Ai
__device__ inline void bilinear_coeff(float dtv, float Ar, float Ai,
                                      float& are, float& aim, float& bsr, float& bsi) {
  float dAr = dtv * Ar, dAi = dtv * Ai;
  float denr = 1.f - 0.5f * dAr, deni = -0.5f * dAi;
  float inv  = 1.f / (denr * denr + deni * deni);
  float numr = 1.f + 0.5f * dAr, numi = 0.5f * dAi;
  are = (numr * denr + numi * deni) * inv;      // a = (1+dA/2)/(1-dA/2)
  aim = (numi * denr - numr * deni) * inv;
  bsr = dtv * denr * inv;                       // bscale = dt/(1-dA/2)
  bsi = -dtv * deni * inv;
}

// A-operand loader: row-major bf16 activations -> WMMA 16x16x32 A layout (ISA 7.12.2)
__device__ inline v16bf load_a(const unsigned short* __restrict__ A, int lane, int ldk, int k0) {
  int m  = lane & 15;
  int kh = (lane >> 4) << 3;        // lanes 0-15: K 0..7/16..23 ; lanes 16-31: 8..15/24..31
  const unsigned short* p = A + (size_t)m * ldk + k0 + kh;
  uint4 lo = *(const uint4*)(p);
  uint4 hi = *(const uint4*)(p + 16);
  v8bf l8 = __builtin_bit_cast(v8bf, lo);
  v8bf h8 = __builtin_bit_cast(v8bf, hi);
  v16bf a;
#pragma unroll
  for (int j = 0; j < 8; ++j) { a[j] = l8[j]; a[8 + j] = h8[j]; }
  return a;
}

// B-operand loader from pre-packed layout: lane's 16 bf16 are contiguous (32 B)
__device__ inline v16bf load_bp(const unsigned short* __restrict__ Bp, int lane) {
  const unsigned short* q = Bp + lane * 16;
  uint4 lo = *(const uint4*)(q);
  uint4 hi = *(const uint4*)(q + 8);
  v8bf l8 = __builtin_bit_cast(v8bf, lo);
  v8bf h8 = __builtin_bit_cast(v8bf, hi);
  v16bf b;
#pragma unroll
  for (int j = 0; j < 8; ++j) { b[j] = l8[j]; b[8 + j] = h8[j]; }
  return b;
}

// ---------------- kernel 0: layernorm (+ flip for backward dir) -> bf16 ----------------
__global__ void ln_kernel(const float* __restrict__ x,
                          const float* __restrict__ wf, const float* __restrict__ bfw,
                          const float* __restrict__ wb, const float* __restrict__ bbw,
                          unsigned short* __restrict__ xnbf) {
  int row = blockIdx.x;            // b*SEQ + l
  int dir = blockIdx.y;
  int tid = threadIdx.x;           // 0..255
  int b = row / SEQ, l = row % SEQ;
  int lsrc = dir ? (SEQ - 1 - l) : l;
  float v = x[((size_t)b * SEQ + lsrc) * DMODEL + tid];
  __shared__ float s1[DMODEL], s2[DMODEL];
  s1[tid] = v; s2[tid] = v * v;
  __syncthreads();
  for (int s = DMODEL / 2; s > 0; s >>= 1) {
    if (tid < s) { s1[tid] += s1[tid + s]; s2[tid] += s2[tid + s]; }
    __syncthreads();
  }
  float mu  = s1[0] * (1.f / DMODEL);
  float var = s2[0] * (1.f / DMODEL) - mu * mu;
  float inv = rsqrtf(var + 1e-5f);
  const float* w  = dir ? wb  : wf;
  const float* bi = dir ? bbw : bfw;
  float yn = (v - mu) * inv * w[tid] + bi[tid];
  xnbf[((size_t)dir * ROWS + row) * DMODEL + tid] = f2bf(yn);
}

// ---------------- kernel 1: pack weight matrix into WMMA B-register layout ----------------
// out[((dir*NTP + nt)*KC + kc)*512 + lane*16 + j], j=0..7 -> K=kc*32+kh+j,
// j=8..15 -> K=kc*32+16+kh+j; n = nt*16 + (lane&15); kh = (lane>>4)*8. Zero-fill n >= N.
__global__ void __launch_bounds__(32)
pack_b_kernel(const float* __restrict__ W0, const float* __restrict__ W1,
              int K, int N, unsigned short* __restrict__ out) {
  int nt = blockIdx.x, kc = blockIdx.y, dir = blockIdx.z;
  int NTP = gridDim.x, KC = gridDim.y;
  int lane = threadIdx.x;
  const float* W = dir ? W1 : W0;
  int n  = nt * 16 + (lane & 15);
  int kh = (lane >> 4) << 3;
  unsigned short vals[16];
#pragma unroll
  for (int j = 0; j < 8; ++j) {
    vals[j]     = (n < N) ? f2bf(W[(size_t)(kc * 32 + kh + j)      * N + n]) : (unsigned short)0;
    vals[8 + j] = (n < N) ? f2bf(W[(size_t)(kc * 32 + 16 + kh + j) * N + n]) : (unsigned short)0;
  }
  unsigned int u[8];
#pragma unroll
  for (int k = 0; k < 8; ++k)
    u[k] = (unsigned int)vals[2 * k] | ((unsigned int)vals[2 * k + 1] << 16);
  unsigned short* dst = out + (((size_t)dir * NTP + nt) * KC + kc) * 512 + (size_t)lane * 16;
  ((uint4*)dst)[0] = make_uint4(u[0], u[1], u[2], u[3]);
  ((uint4*)dst)[1] = make_uint4(u[4], u[5], u[6], u[7]);
}

// ---------------- kernel 2: in-projection GEMM (WMMA bf16, 4 n-tiles/wave) ----------------
__global__ void __launch_bounds__(32)
gemm_in_kernel(const unsigned short* __restrict__ xnbf,
               const unsigned short* __restrict__ winp,
               float* __restrict__ zx) {
  int lane = threadIdx.x;
  int ng = blockIdx.x;             // group of NTW n-tiles
  int mt = blockIdx.y, dir = blockIdx.z;
  const unsigned short* A  = xnbf + ((size_t)dir * ROWS + (size_t)mt * 16) * DMODEL;
  const unsigned short* Bp = winp + (size_t)dir * NTP_IN * KC_IN * 512;
  v8f acc[NTW] = {};
#pragma unroll
  for (int kc = 0; kc < KC_IN; ++kc) {
    v16bf a = load_a(A, lane, DMODEL, kc * 32);
    __builtin_prefetch(A + (size_t)(lane & 15) * DMODEL + kc * 32 + 32, 0, 1);
#pragma unroll
    for (int i = 0; i < NTW; ++i) {
      int nt = ng * NTW + i;
      v16bf b = load_bp(Bp + ((size_t)nt * KC_IN + kc) * 512, lane);
      acc[i] = __builtin_amdgcn_wmma_f32_16x16x32_bf16(false, a, false, b, (short)0, acc[i],
                                                       false, false);
    }
  }
  int mbase = mt * 16 + ((lane >> 4) << 3);
#pragma unroll
  for (int i = 0; i < NTW; ++i) {
    int n = (ng * NTW + i) * 16 + (lane & 15);
    if (n < NPROJ) {
#pragma unroll
      for (int r = 0; r < 8; ++r)
        zx[((size_t)dir * ROWS + mbase + r) * NPROJ + n] = acc[i][r];
    }
  }
}

// ---------------- kernel 3: causal conv + SiLU + split + dt/softplus + v ----------------
__global__ void conv_kernel(const float* __restrict__ zx, Params pf, Params pb,
                            float* __restrict__ ubuf, float* __restrict__ Bmb,
                            float* __restrict__ Cmb, float* __restrict__ dtb,
                            float* __restrict__ vbuf) {
  int l = blockIdx.x, b = blockIdx.y, dir = blockIdx.z;
  const Params P = dir ? pb : pf;
  int ch = threadIdx.x;                       // 0..575
  size_t row = (size_t)dir * ROWS + (size_t)b * SEQ + l;
  __shared__ float su[DINNER];
  float acc = P.convb[ch];
#pragma unroll
  for (int k = 0; k < CONVK; ++k) {
    int ls = l - (CONVK - 1) + k;
    if (ls >= 0)
      acc += P.convw[k * CONVCH + ch] *
             zx[((size_t)dir * ROWS + (size_t)b * SEQ + ls) * NPROJ + DINNER + ch];
  }
  float s = siluf(acc);
  if (ch < DINNER)                { ubuf[row * DINNER + ch] = s; su[ch] = s; }
  else if (ch < DINNER + DSTATE)  Bmb[row * DSTATE + (ch - DINNER)] = s;
  else                            Cmb[row * DSTATE + (ch - DINNER - DSTATE)] = s;
  __syncthreads();
  if (ch < NHEADS * RANK) {                   // v = u @ W_mr_in (per head, R=4)
    int h = ch >> 2, r = ch & 3;
    float a = 0.f;
#pragma unroll 8
    for (int p = 0; p < HEADD; ++p)
      a += su[h * HEADD + p] * P.Wmri[(h * HEADD + p) * RANK + r];
    vbuf[row * NHEADS * RANK + ch] = a;
  }
  if (ch >= 64 && ch < 64 + NHEADS) {         // softplus(dt + bias)
    int h = ch - 64;
    float d = zx[row * NPROJ + DINNER + CONVCH + h] + P.dtb[h];
    dtb[row * NHEADS + h] = (d > 20.f) ? d : log1pf(__expf(d));
  }
}

// ---------------- kernel 4: scan pass A — chunk-local scan (h0 = 0) ----------------
__global__ void __launch_bounds__(32)
scanA_kernel(const float* __restrict__ dtb, const float* __restrict__ Bmb,
             const float* __restrict__ Cmb, const float* __restrict__ vbuf,
             Params pf, Params pb,
             float* __restrict__ ybuf, float* __restrict__ Hc, float* __restrict__ Pc) {
  int c  = blockIdx.x;
  int bh = blockIdx.y; int b = bh / NHEADS, h = bh % NHEADS;
  int dir = blockIdx.z;
  const Params P = dir ? pb : pf;
  int n = threadIdx.x;                        // state index across lanes
  float Ar = -__expf(P.Alog[h * DSTATE + n]);
  float Ai = P.Aim[h * DSTATE + n];
  float hr[RANK] = {0,0,0,0}, hi[RANK] = {0,0,0,0};
  float pr = 1.f, pi = 0.f;                   // cumulative decay product
  size_t base = (size_t)dir * ROWS + (size_t)b * SEQ + (size_t)c * CLEN;
  for (int t = 0; t < CLEN; ++t) {
    size_t row = base + t;
    float dtv = dtb[row * NHEADS + h];
    float are, aim, bsr, bsi;
    bilinear_coeff(dtv, Ar, Ai, are, aim, bsr, bsi);
    float Bn = Bmb[row * DSTATE + n];
    float bbr = bsr * Bn, bbi = bsi * Bn;
    float Cn = Cmb[row * DSTATE + n];
#pragma unroll
    for (int r = 0; r < RANK; ++r) {
      float vr = vbuf[row * NHEADS * RANK + h * RANK + r];
      float nr = are * hr[r] - aim * hi[r] + bbr * vr;
      float ni = are * hi[r] + aim * hr[r] + bbi * vr;
      hr[r] = nr; hi[r] = ni;
      float yp = nr * Cn;                     // y_t = Re(C . h)
#pragma unroll
      for (int off = 16; off > 0; off >>= 1) yp += __shfl_xor(yp, off, 32);
      if (n == 0) ybuf[row * NHEADS * RANK + h * RANK + r] = yp;
    }
    float npr = pr * are - pi * aim;
    float npi = pr * aim + pi * are;
    pr = npr; pi = npi;
  }
  size_t cid = (((size_t)dir * BNB + b) * NHEADS + h) * NCHUNK + c;
#pragma unroll
  for (int r = 0; r < RANK; ++r) {
    Hc[cid * (DSTATE * RANK * 2) + (n * RANK + r) * 2 + 0] = hr[r];
    Hc[cid * (DSTATE * RANK * 2) + (n * RANK + r) * 2 + 1] = hi[r];
  }
  Pc[cid * (DSTATE * 2) + n * 2 + 0] = pr;
  Pc[cid * (DSTATE * 2) + n * 2 + 1] = pi;
}

// ---------------- kernel 5: scan pass B — propagate chunk initial states ----------------
__global__ void __launch_bounds__(32)
scanB_kernel(const float* __restrict__ Hc, const float* __restrict__ Pc,
             float* __restrict__ H0) {
  int h = blockIdx.x, b = blockIdx.y, dir = blockIdx.z;
  int n = threadIdx.x;
  float h0r[RANK] = {0,0,0,0}, h0i[RANK] = {0,0,0,0};
  size_t sid = ((size_t)dir * BNB + b) * NHEADS + h;
  for (int c = 0; c < NCHUNK; ++c) {
    size_t cid = sid * NCHUNK + c;
#pragma unroll
    for (int r = 0; r < RANK; ++r) {
      H0[cid * (DSTATE * RANK * 2) + (n * RANK + r) * 2 + 0] = h0r[r];
      H0[cid * (DSTATE * RANK * 2) + (n * RANK + r) * 2 + 1] = h0i[r];
    }
    float pr = Pc[cid * (DSTATE * 2) + n * 2 + 0];
    float pi = Pc[cid * (DSTATE * 2) + n * 2 + 1];
#pragma unroll
    for (int r = 0; r < RANK; ++r) {
      float cr = Hc[cid * (DSTATE * RANK * 2) + (n * RANK + r) * 2 + 0];
      float ci = Hc[cid * (DSTATE * RANK * 2) + (n * RANK + r) * 2 + 1];
      float nr = pr * h0r[r] - pi * h0i[r] + cr;
      float ni = pr * h0i[r] + pi * h0r[r] + ci;
      h0r[r] = nr; h0i[r] = ni;
    }
  }
}

// ---------------- kernel 6: scan pass C — add h0 correction to outputs ----------------
__global__ void __launch_bounds__(32)
scanC_kernel(const float* __restrict__ dtb, const float* __restrict__ Cmb,
             const float* __restrict__ H0, Params pf, Params pb,
             float* __restrict__ ybuf) {
  int c = blockIdx.x;
  if (c == 0) return;                          // chunk 0 has h0 == 0
  int bh = blockIdx.y; int b = bh / NHEADS, h = bh % NHEADS;
  int dir = blockIdx.z;
  const Params P = dir ? pb : pf;
  int n = threadIdx.x;
  float Ar = -__expf(P.Alog[h * DSTATE + n]);
  float Ai = P.Aim[h * DSTATE + n];
  size_t cid = (((size_t)dir * BNB + b) * NHEADS + h) * NCHUNK + c;
  float h0r[RANK], h0i[RANK];
#pragma unroll
  for (int r = 0; r < RANK; ++r) {
    h0r[r] = H0[cid * (DSTATE * RANK * 2) + (n * RANK + r) * 2 + 0];
    h0i[r] = H0[cid * (DSTATE * RANK * 2) + (n * RANK + r) * 2 + 1];
  }
  float pr = 1.f, pi = 0.f;
  size_t base = (size_t)dir * ROWS + (size_t)b * SEQ + (size_t)c * CLEN;
  for (int t = 0; t < CLEN; ++t) {
    size_t row = base + t;
    float dtv = dtb[row * NHEADS + h];
    float are, aim, bsr, bsi;
    bilinear_coeff(dtv, Ar, Ai, are, aim, bsr, bsi);
    float npr = pr * are - pi * aim;
    float npi = pr * aim + pi * are;
    pr = npr; pi = npi;                        // prod_{s<=t} a_s
    float Cn = Cmb[row * DSTATE + n];
#pragma unroll
    for (int r = 0; r < RANK; ++r) {
      float g = (pr * h0r[r] - pi * h0i[r]) * Cn;
#pragma unroll
      for (int off = 16; off > 0; off >>= 1) g += __shfl_xor(g, off, 32);
      if (n == 0) ybuf[row * NHEADS * RANK + h * RANK + r] += g;
    }
  }
}

// ---------------- kernel 7: W_mr_out proj + D skip + gate + RMS norm -> bf16 ----------------
__global__ void gate_norm_kernel(const float* __restrict__ ybuf, const float* __restrict__ ubuf,
                                 const float* __restrict__ zx, Params pf, Params pb,
                                 unsigned short* __restrict__ yibf) {
  int l = blockIdx.x, b = blockIdx.y, dir = blockIdx.z;
  const Params P = dir ? pb : pf;
  int c = threadIdx.x;                         // 0..511
  int h = c >> 6, p = c & 63;
  size_t row = (size_t)dir * ROWS + (size_t)b * SEQ + l;
  float yv = 0.f;
#pragma unroll
  for (int r = 0; r < RANK; ++r)
    yv += ybuf[row * NHEADS * RANK + h * RANK + r] * P.Wmro[(h * RANK + r) * HEADD + p];
  yv += ubuf[row * DINNER + c] * P.Dsk[h];
  float z = zx[row * NPROJ + c];
  float g = yv * siluf(z);
  __shared__ float ss[DINNER];
  ss[c] = g * g;
  __syncthreads();
  for (int s = DINNER / 2; s > 0; s >>= 1) {
    if (c < s) ss[c] += ss[c + s];
    __syncthreads();
  }
  float rms = rsqrtf(ss[0] * (1.f / DINNER) + 1e-5f);
  yibf[row * DINNER + c] = f2bf(g * rms * P.normw[c]);
}

// ---------------- kernel 8: out-projection GEMM (WMMA bf16, 4 n-tiles/wave) ----------------
__global__ void __launch_bounds__(32)
gemm_out_kernel(const unsigned short* __restrict__ yibf,
                const unsigned short* __restrict__ woutp,
                float* __restrict__ out) {
  int lane = threadIdx.x;
  int ng = blockIdx.x;             // group of NTW n-tiles (4 groups cover 256 cols)
  int mt = blockIdx.y, dir = blockIdx.z;
  const unsigned short* A  = yibf + ((size_t)dir * ROWS + (size_t)mt * 16) * DINNER;
  const unsigned short* Bp = woutp + (size_t)dir * NT_OUT * KC_OUT * 512;
  v8f acc[NTW] = {};
#pragma unroll
  for (int kc = 0; kc < KC_OUT; ++kc) {
    v16bf a = load_a(A, lane, DINNER, kc * 32);
    __builtin_prefetch(A + (size_t)(lane & 15) * DINNER + kc * 32 + 32, 0, 1);
#pragma unroll
    for (int i = 0; i < NTW; ++i) {
      int nt = ng * NTW + i;
      v16bf b = load_bp(Bp + ((size_t)nt * KC_OUT + kc) * 512, lane);
      acc[i] = __builtin_amdgcn_wmma_f32_16x16x32_bf16(false, a, false, b, (short)0, acc[i],
                                                       false, false);
    }
  }
  int mbase = mt * 16 + ((lane >> 4) << 3);
#pragma unroll
  for (int i = 0; i < NTW; ++i) {
    int n = (ng * NTW + i) * 16 + (lane & 15);
#pragma unroll
    for (int r = 0; r < 8; ++r) {
      int m = mbase + r;
      int b_ = m / SEQ, l = m % SEQ;
      int lo = dir ? (SEQ - 1 - l) : l;        // un-flip backward direction
      out[((size_t)b_ * SEQ + lo) * (2 * DMODEL) + dir * DMODEL + n] = acc[i][r];
    }
  }
}

// ---------------- host launcher ----------------
extern "C" void kernel_launch(void* const* d_in, const int* in_sizes, int n_in,
                              void* d_out, int out_size, void* d_ws, size_t ws_size,
                              hipStream_t stream) {
  (void)in_sizes; (void)n_in; (void)out_size; (void)ws_size;
  const float* x   = (const float*)d_in[0];
  const float* nfw = (const float*)d_in[1];
  const float* nfb = (const float*)d_in[2];
  const float* nbw = (const float*)d_in[3];
  const float* nbb = (const float*)d_in[4];
  Params pf { (const float*)d_in[5],  (const float*)d_in[6],  (const float*)d_in[7],
              (const float*)d_in[8],  (const float*)d_in[9],  (const float*)d_in[10],
              (const float*)d_in[11], (const float*)d_in[12], (const float*)d_in[13],
              (const float*)d_in[14], (const float*)d_in[15] };
  Params pb { (const float*)d_in[16], (const float*)d_in[17], (const float*)d_in[18],
              (const float*)d_in[19], (const float*)d_in[20], (const float*)d_in[21],
              (const float*)d_in[22], (const float*)d_in[23], (const float*)d_in[24],
              (const float*)d_in[25], (const float*)d_in[26] };

  char* w = (char*)d_ws; size_t off = 0;
  auto carve = [&](size_t bytes) -> void* {
    void* p = w + off;
    off = (off + bytes + 255) & ~(size_t)255;
    return p;
  };
  unsigned short* xnbf  = (unsigned short*)carve((size_t)2 * ROWS * DMODEL * 2);
  unsigned short* winp  = (unsigned short*)carve((size_t)2 * NTP_IN * KC_IN * 512 * 2);
  unsigned short* woutp = (unsigned short*)carve((size_t)2 * NT_OUT * KC_OUT * 512 * 2);
  float* zx   = (float*)carve((size_t)2 * ROWS * NPROJ * 4);
  float* ubuf = (float*)carve((size_t)2 * ROWS * DINNER * 4);
  float* Bmb  = (float*)carve((size_t)2 * ROWS * DSTATE * 4);
  float* Cmb  = (float*)carve((size_t)2 * ROWS * DSTATE * 4);
  float* dtb  = (float*)carve((size_t)2 * ROWS * NHEADS * 4);
  float* vbuf = (float*)carve((size_t)2 * ROWS * NHEADS * RANK * 4);
  float* ybuf = (float*)carve((size_t)2 * ROWS * NHEADS * RANK * 4);
  float* Hc   = (float*)carve((size_t)2 * BNB * NHEADS * NCHUNK * DSTATE * RANK * 2 * 4);
  float* Pc   = (float*)carve((size_t)2 * BNB * NHEADS * NCHUNK * DSTATE * 2 * 4);
  float* H0   = (float*)carve((size_t)2 * BNB * NHEADS * NCHUNK * DSTATE * RANK * 2 * 4);
  unsigned short* yibf = (unsigned short*)carve((size_t)2 * ROWS * DINNER * 2);

  ln_kernel<<<dim3(ROWS, 2), DMODEL, 0, stream>>>(x, nfw, nfb, nbw, nbb, xnbf);
  pack_b_kernel<<<dim3(NTP_IN, KC_IN, 2), 32, 0, stream>>>(pf.Win, pb.Win,
                                                           DMODEL, NPROJ, winp);
  pack_b_kernel<<<dim3(NT_OUT, KC_OUT, 2), 32, 0, stream>>>(pf.Wout, pb.Wout,
                                                            DINNER, DMODEL, woutp);
  gemm_in_kernel<<<dim3(NTP_IN / NTW, MTILES, 2), 32, 0, stream>>>(xnbf, winp, zx);
  conv_kernel<<<dim3(SEQ, BNB, 2), CONVCH, 0, stream>>>(zx, pf, pb, ubuf, Bmb, Cmb, dtb, vbuf);
  scanA_kernel<<<dim3(NCHUNK, BNB * NHEADS, 2), 32, 0, stream>>>(dtb, Bmb, Cmb, vbuf, pf, pb,
                                                                 ybuf, Hc, Pc);
  scanB_kernel<<<dim3(NHEADS, BNB, 2), 32, 0, stream>>>(Hc, Pc, H0);
  scanC_kernel<<<dim3(NCHUNK, BNB * NHEADS, 2), 32, 0, stream>>>(dtb, Cmb, H0, pf, pb, ybuf);
  gate_norm_kernel<<<dim3(SEQ, BNB, 2), DINNER, 0, stream>>>(ybuf, ubuf, zx, pf, pb, yibf);
  gemm_out_kernel<<<dim3(NT_OUT / NTW, MTILES, 2), 32, 0, stream>>>(yibf, woutp, (float*)d_out);
}
